// HybridLayer_50929722196045
// MI455X (gfx1250) — matrix operand order repaired and checked by reading, not compiled, not run on hardware
//
#include <hip/hip_runtime.h>
#include <hip/hip_bf16.h>
#include <math.h>

typedef __attribute__((ext_vector_type(16))) __bf16 bf16x16;
typedef __attribute__((ext_vector_type(8)))  __bf16 bf16x8;
typedef __attribute__((ext_vector_type(8)))  float  f32x8;

#define B_     2
#define S_     2048
#define D_     1024
#define INNER_ 2048
#define KSEL   1024
#define NTOK   2048      // B_*KSEL
#define NEXP   8
#define HID    4096
#define EPSV   1e-5f

// ---------------------------------------------------------------- GEMM (bf16 WMMA)
// BM=128, BN=64, BK=64; 8 waves (4 x 2), each wave owns 32x32 output
// -> 8 v_wmma per LDS stage per wave.
#define BM 128
#define BN 64
#define BK 64
#define LDSS (BK + 8)   // 72 bf16 = 144B row stride (multiple of 16B)

__device__ inline bf16x16 frag_A(const __bf16* rowbase, int h) {
  // 16-bit A 16x32 layout: lane half h holds K [h*8 .. h*8+7] and [16+h*8 .. +7]
  union { bf16x16 v; bf16x8 p[2]; } u;
  u.p[0] = *(const bf16x8*)(rowbase + h * 8);
  u.p[1] = *(const bf16x8*)(rowbase + 16 + h * 8);
  return u.v;
}
__device__ inline bf16x16 frag_B(const __bf16* colbase, int h) {
  // 16-bit B 32x16 layout: lane half h holds K [h*8 .. h*8+7] and [16+h*8 .. +7]
  union { bf16x16 v; bf16x8 p[2]; } u;
  u.p[0] = *(const bf16x8*)(colbase + h * 8);
  u.p[1] = *(const bf16x8*)(colbase + 16 + h * 8);
  return u.v;
}

#define WMMA_BF16(a, b, c) \
  __builtin_amdgcn_wmma_f32_16x16x32_bf16(false, (a), false, (b), (short)0, (c), false, false)

// EPI: 0 plain store f32 | 1 conv-scale (n<INNER_) store f32 | 2 += into f32
//      3 gelu(bias+acc) -> bf16  | 4 moe: Cf[rowIdx[m]] += gate*(acc+bias)
template <int EPI, bool AIND>
__global__ __launch_bounds__(256) void gemm_bf16_kernel(
    const __bf16* __restrict__ A, const float* __restrict__ Bw,
    int M, int N, int K, int ldA,
    const int* __restrict__ rowIdx, const int* __restrict__ rowCnt,
    float* __restrict__ Cf, __bf16* __restrict__ Cb,
    const float* __restrict__ bias, const float* __restrict__ scale,
    const float* __restrict__ gate, int expert) {
  __shared__ __bf16 As[BM][LDSS];
  __shared__ __bf16 Bs[BN][LDSS];

  const int rowBase = blockIdx.y * BM;
  const int colBase = blockIdx.x * BN;
  const int limit = rowCnt ? *rowCnt : M;
  if (rowBase >= limit) return;

  const int tid  = threadIdx.x;
  const int wave = tid >> 5;
  const int lane = tid & 31;
  const int wm = wave & 3;     // 4 wave rows  -> 128 M (32 each)
  const int wn = wave >> 2;    // 2 wave cols  -> 64 N  (32 each)
  const int lr = lane & 15;
  const int lh = lane >> 4;

  f32x8 acc[2][2];
#pragma unroll
  for (int a = 0; a < 2; ++a)
#pragma unroll
    for (int b = 0; b < 2; ++b)
#pragma unroll
      for (int i = 0; i < 8; ++i) acc[a][b][i] = 0.f;

  // A staging decomposition: thread -> (row, 32-elt chunk)
  const int ar = tid >> 1;            // 0..127
  const int ak = (tid & 1) * 32;      // 0 or 32
  // B staging decomposition: thread -> (col, 16 k values)
  const int bn = tid & 63;
  const int bk = (tid >> 6) * 16;

  for (int k0 = 0; k0 < K; k0 += BK) {
    __syncthreads();
    { // A tile: 128x64 bf16; 64B per thread via async global->LDS (CDNA5 path)
      int gr = rowBase + ar;
      if (gr < limit) {
        int grow = AIND ? rowIdx[gr] : gr;
        unsigned lds_a = (unsigned)(uintptr_t)&As[ar][ak];
        unsigned long long ga =
            (unsigned long long)(uintptr_t)(A + (size_t)grow * ldA + k0 + ak);
        asm volatile(
            "global_load_async_to_lds_b128 %0, %1, off\n\t"
            "global_load_async_to_lds_b128 %0, %1, off offset:16\n\t"
            "global_load_async_to_lds_b128 %0, %1, off offset:32\n\t"
            "global_load_async_to_lds_b128 %0, %1, off offset:48"
            :: "v"(lds_a), "v"(ga) : "memory");
      } else {
#pragma unroll
        for (int q = 0; q < 32; ++q) As[ar][ak + q] = (__bf16)0.f;
      }
    }
    { // B tile: transpose-stage 64(K) x 64(N) f32 -> Bs[n][k] bf16
      const float* src = Bw + (size_t)(k0 + bk) * N + colBase + bn;
#pragma unroll
      for (int q = 0; q < 16; ++q)
        Bs[bn][bk + q] = (__bf16)src[(size_t)q * N];
      if (k0 + BK < K) __builtin_prefetch(src + (size_t)BK * N, 0, 0);
    }
    asm volatile("s_wait_asynccnt 0x0" ::: "memory");
    __syncthreads();

#pragma unroll
    for (int ks = 0; ks < 2; ++ks) {
      bf16x16 a0 = frag_A(&As[wm * 32 + lr][ks * 32], lh);
      bf16x16 a1 = frag_A(&As[wm * 32 + 16 + lr][ks * 32], lh);
      bf16x16 b0 = frag_B(&Bs[wn * 32 + lr][ks * 32], lh);
      bf16x16 b1 = frag_B(&Bs[wn * 32 + 16 + lr][ks * 32], lh);
      acc[0][0] = WMMA_BF16(a0, b0, acc[0][0]);
      acc[0][1] = WMMA_BF16(a0, b1, acc[0][1]);
      acc[1][0] = WMMA_BF16(a1, b0, acc[1][0]);
      acc[1][1] = WMMA_BF16(a1, b1, acc[1][1]);
    }
  }

#pragma unroll
  for (int mi = 0; mi < 2; ++mi) {
    const int mB = rowBase + wm * 32 + mi * 16 + lh * 8;
#pragma unroll
    for (int j = 0; j < 8; ++j) {
      int m = mB + j;
      if (m >= limit) continue;
#pragma unroll
      for (int nn = 0; nn < 2; ++nn) {
        float v = acc[mi][nn][j];
        int n = colBase + wn * 32 + nn * 16 + lr;
        n += nn ? 0 : 0;  // (kept simple; nn already in index below)
        n = colBase + wn * 32 + nn * 16 + lr;
        if constexpr (EPI == 0) {
          Cf[(size_t)m * N + n] = v;
        } else if constexpr (EPI == 1) {
          if (n < INNER_) v = v * scale[n] + bias[n];
          Cf[(size_t)m * N + n] = v;
        } else if constexpr (EPI == 2) {
          Cf[(size_t)m * N + n] += v;
        } else if constexpr (EPI == 3) {
          float t = v + bias[n];
          float g = 0.5f * t * (1.0f + erff(t * 0.70710678118654752f));
          Cb[(size_t)m * N + n] = (__bf16)g;
        } else {
          int tk = rowIdx[m];
          Cf[(size_t)tk * N + n] += gate[tk * NEXP + expert] * (v + bias[n]);
        }
      }
    }
  }
}

// ---------------------------------------------------------------- router scores
__global__ __launch_bounds__(256) void router_scores_kernel(
    const float* __restrict__ x, const float* __restrict__ w1,
    const float* __restrict__ b1, const float* __restrict__ w2,
    const float* __restrict__ b2, float* __restrict__ scores) {
  __shared__ float xs[D_];
  __shared__ float red[256];
  int n = blockIdx.x, tid = threadIdx.x;
  const float* xr = x + (size_t)n * D_;
  for (int j = tid; j < D_; j += 256) xs[j] = xr[j];
  __syncthreads();
  float h = b1[tid];
  for (int j = 0; j < D_; ++j) h = fmaf(xs[j], w1[(size_t)j * 256 + tid], h);
  h = h > 0.f ? h : 0.f;
  red[tid] = h * w2[tid];
  __syncthreads();
  for (int s = 128; s > 0; s >>= 1) {
    if (tid < s) red[tid] += red[tid + s];
    __syncthreads();
  }
  if (tid == 0) scores[n] = red[0] + b2[0];   // TEMP == 1
}

// ---------------------------------------------------------------- top-k (bitonic)
__global__ __launch_bounds__(1024) void topk_kernel(
    const float* __restrict__ scores, int* __restrict__ sidx,
    float* __restrict__ wrank) {
  __shared__ unsigned long long keys[S_];
  __shared__ float red[1024];
  __shared__ float smax;
  __shared__ unsigned idxs[KSEL];
  int b = blockIdx.x, tid = threadIdx.x;

  for (int i = tid; i < S_; i += 1024) {
    unsigned u = __float_as_uint(scores[b * S_ + i]);
    u ^= (u & 0x80000000u) ? 0xFFFFFFFFu : 0x80000000u;  // ascending-sortable
    unsigned du = ~u;                                    // descending
    keys[i] = ((unsigned long long)du << 32) | (unsigned)i;
  }
  for (int k = 2; k <= S_; k <<= 1)
    for (int j = k >> 1; j > 0; j >>= 1) {
      __syncthreads();
      for (int i = tid; i < S_; i += 1024) {
        int ixj = i ^ j;
        if (ixj > i) {
          bool up = ((i & k) == 0);
          unsigned long long a = keys[i], c = keys[ixj];
          if ((a > c) == up) { keys[i] = c; keys[ixj] = a; }
        }
      }
    }
  __syncthreads();
  // rank tid score (descending order = ascending key)
  unsigned du = (unsigned)(keys[tid] >> 32);
  unsigned u = ~du;
  unsigned fb = (u & 0x80000000u) ? (u ^ 0x80000000u) : (u ^ 0xFFFFFFFFu);
  float sc = __uint_as_float(fb);
  if (tid == 0) smax = sc;
  __syncthreads();
  float ev = expf(sc - smax);
  red[tid] = ev;
  __syncthreads();
  for (int s = 512; s > 0; s >>= 1) {
    if (tid < s) red[tid] += red[tid + s];
    __syncthreads();
  }
  wrank[b * KSEL + tid] = ev / red[0];   // softmax weight BY RANK (reference quirk)
  idxs[tid] = (unsigned)(keys[tid] & 0xFFFFFFFFu);
  __syncthreads();
  for (int k = 2; k <= KSEL; k <<= 1)
    for (int j = k >> 1; j > 0; j >>= 1) {
      int ixj = tid ^ j;
      if (ixj > tid) {
        bool up = ((tid & k) == 0);
        unsigned a = idxs[tid], c = idxs[ixj];
        if ((a > c) == up) { idxs[tid] = c; idxs[ixj] = a; }
      }
      __syncthreads();
    }
  sidx[b * KSEL + tid] = (int)idxs[tid];
}

// ---------------------------------------------------------------- LN (opt gather)
template <bool GATHER>
__global__ __launch_bounds__(256) void ln_kernel(
    const float* __restrict__ src, const int* __restrict__ sidx,
    float* __restrict__ tokOut, __bf16* __restrict__ nOut,
    const float* __restrict__ lw, const float* __restrict__ lb) {
  __shared__ float red[256];
  int j = blockIdx.x, tid = threadIdx.x;
  const float* row;
  float* trow = nullptr;
  if (GATHER) {
    int b = j >> 10, jj = j & (KSEL - 1);
    int s = sidx[b * KSEL + jj];
    row = src + ((size_t)b * S_ + s) * D_;
    trow = tokOut + (size_t)j * D_;
  } else {
    row = src + (size_t)j * D_;
  }
  float p = 0.f;
  for (int t = tid; t < D_; t += 256) {
    float v = row[t];
    if (GATHER) trow[t] = v;
    p += v;
  }
  red[tid] = p;
  __syncthreads();
  for (int s = 128; s > 0; s >>= 1) { if (tid < s) red[tid] += red[tid + s]; __syncthreads(); }
  float mu = red[0] / D_;
  __syncthreads();
  p = 0.f;
  for (int t = tid; t < D_; t += 256) { float d = row[t] - mu; p += d * d; }
  red[tid] = p;
  __syncthreads();
  for (int s = 128; s > 0; s >>= 1) { if (tid < s) red[tid] += red[tid + s]; __syncthreads(); }
  float rs = rsqrtf(red[0] / D_ + EPSV);
  for (int t = tid; t < D_; t += 256)
    nOut[(size_t)j * D_ + t] = (__bf16)((row[t] - mu) * rs * lw[t] + lb[t]);
}

// ---------------------------------------------------------------- small preps
__global__ void prep_kernel(const float* __restrict__ xproj,
                            const float* __restrict__ convw,
                            float* __restrict__ wB, float* __restrict__ wC,
                            float* __restrict__ cw3) {
  int j = blockIdx.x * blockDim.x + threadIdx.x;
  if (j < INNER_) {
    float s = 0.f;
    for (int c = 16; c < 32; ++c) s += xproj[j * 33 + c];
    wB[j] = s;
    wC[j] = xproj[j * 33 + 32];
    cw3[j] = convw[j * 4 + 3];
  }
}

__global__ __launch_bounds__(256) void rowdots_kernel(
    const float* __restrict__ UV, const float* __restrict__ wB,
    const float* __restrict__ wC, float* __restrict__ g) {
  __shared__ float r1[256], r2[256];
  int n = blockIdx.x, tid = threadIdx.x;
  const float* u = UV + (size_t)n * (2 * INNER_);
  float p1 = 0.f, p2 = 0.f;
  for (int j = tid; j < INNER_; j += 256) {
    float v = u[j];
    p1 = fmaf(v, wB[j], p1);
    p2 = fmaf(v, wC[j], p2);
  }
  r1[tid] = p1; r2[tid] = p2;
  __syncthreads();
  for (int s = 128; s > 0; s >>= 1) {
    if (tid < s) { r1[tid] += r1[tid + s]; r2[tid] += r2[tid + s]; }
    __syncthreads();
  }
  if (tid == 0) g[n] = r1[0] * r2[0];
}

__global__ void yelem_kernel(const float* __restrict__ UV,
                             const float* __restrict__ g,
                             __bf16* __restrict__ Y) {
  size_t i = (size_t)blockIdx.x * 256 + threadIdx.x;
  int n = (int)(i / INNER_), jj = (int)(i % INNER_);
  const float* row = UV + (size_t)n * 2 * INNER_;
  float u = row[jj], res = row[INNER_ + jj];
  float sil = res / (1.f + expf(-res));
  Y[i] = (__bf16)(u * g[n] * sil);
}

// ---------------------------------------------------------------- MoE router
__global__ __launch_bounds__(256) void moerouter_kernel(
    const __bf16* __restrict__ n2, const float* __restrict__ rw,
    const float* __restrict__ rb, float* __restrict__ gate,
    int* __restrict__ ecnt, int* __restrict__ eidx) {
  __shared__ float red[NEXP][256];
  int n = blockIdx.x, tid = threadIdx.x;
  float acc[NEXP];
#pragma unroll
  for (int e = 0; e < NEXP; ++e) acc[e] = 0.f;
  for (int j = tid; j < D_; j += 256) {
    float xv = (float)n2[(size_t)n * D_ + j];
#pragma unroll
    for (int e = 0; e < NEXP; ++e) acc[e] = fmaf(xv, rw[j * NEXP + e], acc[e]);
  }
  for (int e = 0; e < NEXP; ++e) red[e][tid] = acc[e];
  __syncthreads();
  for (int s = 128; s > 0; s >>= 1) {
    if (tid < s)
      for (int e = 0; e < NEXP; ++e) red[e][tid] += red[e][tid + s];
    __syncthreads();
  }
  if (tid == 0) {
    float lg[NEXP], mx = -1e30f;
    for (int e = 0; e < NEXP; ++e) { lg[e] = red[e][0] + rb[e]; mx = fmaxf(mx, lg[e]); }
    float p[NEXP], sum = 0.f;
    for (int e = 0; e < NEXP; ++e) { p[e] = expf(lg[e] - mx); sum += p[e]; }
    for (int e = 0; e < NEXP; ++e) p[e] /= sum;
    int i1 = 0;
    for (int e = 1; e < NEXP; ++e) if (p[e] > p[i1]) i1 = e;
    int i2 = -1;
    for (int e = 0; e < NEXP; ++e)
      if (e != i1 && (i2 < 0 || p[e] > p[i2])) i2 = e;
    float s2 = p[i1] + p[i2];
    for (int e = 0; e < NEXP; ++e) gate[n * NEXP + e] = 0.f;
    gate[n * NEXP + i1] = p[i1] / s2;
    gate[n * NEXP + i2] = p[i2] / s2;
    int p1 = atomicAdd(&ecnt[i1], 1); eidx[i1 * NTOK + p1] = n;
    int p2 = atomicAdd(&ecnt[i2], 1); eidx[i2 * NTOK + p2] = n;
  }
}

// ---------------------------------------------------------------- misc
__global__ void init_kernel(float* __restrict__ moe_out, int* __restrict__ ecnt) {
  size_t i = (size_t)blockIdx.x * 256 + threadIdx.x;
  if (i < (size_t)NTOK * D_) moe_out[i] = 0.f;
  if (i < NEXP) ecnt[i] = 0;
}

__global__ void copy_kernel(const float* __restrict__ x, float* __restrict__ out) {
  size_t i = (size_t)blockIdx.x * 256 + threadIdx.x;
  if (i < (size_t)B_ * S_ * D_) out[i] = x[i];
}

__global__ __launch_bounds__(256) void scatter_kernel(
    const float* __restrict__ tok, const float* __restrict__ moe_out,
    const int* __restrict__ sidx, const float* __restrict__ wr,
    float* __restrict__ out) {
  int j = blockIdx.x, tid = threadIdx.x;
  int b = j >> 10, jj = j & (KSEL - 1);
  int s = sidx[b * KSEL + jj];
  float wj = wr[b * KSEL + jj];
  float* dst = out + ((size_t)b * S_ + s) * D_;
  const float* t1 = tok + (size_t)j * D_;
  const float* t2 = moe_out + (size_t)j * D_;
  for (int t = tid; t < D_; t += 256) dst[t] = (t1[t] + t2[t]) * wj;
}

__global__ void lb_kernel(const int* __restrict__ ecnt, float* __restrict__ out_lb) {
  if (threadIdx.x == 0 && blockIdx.x == 0) {
    float c[NEXP], sum = 0.f;
    for (int e = 0; e < NEXP; ++e) { c[e] = (float)ecnt[e]; sum += c[e]; }
    float mean = sum / (float)NEXP;
    float var = 0.f;
    for (int e = 0; e < NEXP; ++e) { float d = c[e] - mean; var += d * d; }
    var /= (float)(NEXP - 1);
    out_lb[0] = var / mean * 0.01f;
  }
}

// ---------------------------------------------------------------- launch
extern "C" void kernel_launch(void* const* d_in, const int* in_sizes, int n_in,
                              void* d_out, int out_size, void* d_ws, size_t ws_size,
                              hipStream_t stream) {
  const float* x        = (const float*)d_in[0];
  const float* ln1_w    = (const float*)d_in[1];
  const float* ln1_b    = (const float*)d_in[2];
  const float* ln2_w    = (const float*)d_in[3];
  const float* ln2_b    = (const float*)d_in[4];
  const float* modr_w1  = (const float*)d_in[5];
  const float* modr_b1  = (const float*)d_in[6];
  const float* modr_w2  = (const float*)d_in[7];
  const float* modr_b2  = (const float*)d_in[8];
  const float* m_in_w   = (const float*)d_in[9];
  const float* m_conv_w = (const float*)d_in[10];
  const float* m_conv_b = (const float*)d_in[11];
  const float* m_xproj  = (const float*)d_in[12];
  const float* m_out_w  = (const float*)d_in[13];
  const float* moe_r_w  = (const float*)d_in[14];
  const float* moe_r_b  = (const float*)d_in[15];
  const float* moe_w1   = (const float*)d_in[16];
  const float* moe_b1   = (const float*)d_in[17];
  const float* moe_w2   = (const float*)d_in[18];
  const float* moe_b2   = (const float*)d_in[19];

  float* out = (float*)d_out;
  float* out_lb = out + (size_t)B_ * S_ * D_;

  char* w = (char*)d_ws;
  size_t o = 0;
  auto alloc = [&](size_t bytes) {
    size_t cur = o;
    o = (o + bytes + 255) & ~(size_t)255;
    return (void*)(w + cur);
  };
  float*  UV      = (float*)alloc((size_t)NTOK * 2 * INNER_ * 4);
  float*  tok     = (float*)alloc((size_t)NTOK * D_ * 4);
  __bf16* n1      = (__bf16*)alloc((size_t)NTOK * D_ * 2);
  __bf16* Y       = (__bf16*)alloc((size_t)NTOK * INNER_ * 2);
  __bf16* n2      = (__bf16*)alloc((size_t)NTOK * D_ * 2);
  __bf16* H       = (__bf16*)alloc((size_t)NTOK * HID * 2);
  float*  moe_out = (float*)alloc((size_t)NTOK * D_ * 4);
  float*  scores  = (float*)alloc((size_t)B_ * S_ * 4);
  int*    sidx    = (int*)alloc((size_t)NTOK * 4);
  float*  wrank   = (float*)alloc((size_t)NTOK * 4);
  float*  gvec    = (float*)alloc((size_t)NTOK * 4);
  float*  wB      = (float*)alloc((size_t)INNER_ * 4);
  float*  wC      = (float*)alloc((size_t)INNER_ * 4);
  float*  cw3     = (float*)alloc((size_t)INNER_ * 4);
  float*  gate    = (float*)alloc((size_t)NTOK * NEXP * 4);
  int*    ecnt    = (int*)alloc((size_t)NEXP * 4);
  int*    eidx    = (int*)alloc((size_t)NEXP * NTOK * 4);

  init_kernel<<<(NTOK * D_ + 255) / 256, 256, 0, stream>>>(moe_out, ecnt);
  prep_kernel<<<(INNER_ + 255) / 256, 256, 0, stream>>>(m_xproj, m_conv_w, wB, wC, cw3);
  router_scores_kernel<<<B_ * S_, 256, 0, stream>>>(x, modr_w1, modr_b1, modr_w2,
                                                    modr_b2, scores);
  topk_kernel<<<B_, 1024, 0, stream>>>(scores, sidx, wrank);
  ln_kernel<true><<<NTOK, 256, 0, stream>>>(x, sidx, tok, n1, ln1_w, ln1_b);
  // uv = n1 @ m_in_w ; u-part conv-scaled in epilogue
  gemm_bf16_kernel<1, false><<<dim3(2 * INNER_ / BN, NTOK / BM), 256, 0, stream>>>(
      n1, m_in_w, NTOK, 2 * INNER_, D_, D_, nullptr, nullptr, UV, nullptr,
      m_conv_b, cw3, nullptr, 0);
  rowdots_kernel<<<NTOK, 256, 0, stream>>>(UV, wB, wC, gvec);
  yelem_kernel<<<(NTOK * INNER_) / 256, 256, 0, stream>>>(UV, gvec, Y);
  // tok += y @ m_out_w
  gemm_bf16_kernel<2, false><<<dim3(D_ / BN, NTOK / BM), 256, 0, stream>>>(
      Y, m_out_w, NTOK, D_, INNER_, INNER_, nullptr, nullptr, tok, nullptr,
      nullptr, nullptr, nullptr, 0);
  ln_kernel<false><<<NTOK, 256, 0, stream>>>(tok, nullptr, nullptr, n2, ln2_w, ln2_b);
  moerouter_kernel<<<NTOK, 256, 0, stream>>>(n2, moe_r_w, moe_r_b, gate, ecnt, eidx);
  for (int e = 0; e < NEXP; ++e) {
    gemm_bf16_kernel<3, true><<<dim3(HID / BN, NTOK / BM), 256, 0, stream>>>(
        n2, moe_w1 + (size_t)e * D_ * HID, NTOK, HID, D_, D_,
        eidx + e * NTOK, ecnt + e, nullptr, H, moe_b1 + (size_t)e * HID,
        nullptr, nullptr, e);
    gemm_bf16_kernel<4, false><<<dim3(D_ / BN, NTOK / BM), 256, 0, stream>>>(
        H, moe_w2 + (size_t)e * HID * D_, NTOK, D_, HID, HID,
        eidx + e * NTOK, ecnt + e, moe_out, nullptr, moe_b2 + (size_t)e * D_,
        nullptr, gate, e);
  }
  copy_kernel<<<(B_ * S_ * D_ + 255) / 256, 256, 0, stream>>>(x, out);
  scatter_kernel<<<NTOK, 256, 0, stream>>>(tok, moe_out, sidx, wrank, out);
  lb_kernel<<<1, 32, 0, stream>>>(ecnt, out_lb);
}
